// ParticleNet_19344532701420
// MI455X (gfx1250) — matrix operand order repaired and checked by reading, not compiled, not run on hardware
//
#include <hip/hip_runtime.h>
#include <hip/hip_bf16.h>
#include <stdint.h>

typedef __attribute__((ext_vector_type(16))) _Float16 v16h;
typedef __attribute__((ext_vector_type(8)))  _Float16 v8h;
typedef __attribute__((ext_vector_type(8)))  float    v8f;

#define LDS_STRIDE 40          // 32 halves + 8 pad (80B rows, 16B aligned chunks)
#define BN_EPS_F 1e-5f

// ---- CDNA5 async global->LDS staging (ASYNCcnt path, ISA 08_async_tensor §4)
// vdst = 32-bit LDS byte offset VGPR, vaddr = 64-bit global address, saddr=off.
__device__ __forceinline__ void async_copy_b128(unsigned lds_off, const void* gaddr) {
  asm volatile("global_load_async_to_lds_b128 %0, %1, off"
               :: "v"(lds_off), "v"(gaddr) : "memory");
}
__device__ __forceinline__ void wait_async_zero() {
  asm volatile("s_wait_asynccnt 0" ::: "memory");
}
__device__ __forceinline__ unsigned lds_offset_of(const void* p) {
  // flat LDS addresses carry the wave-relative offset in addr[31:0]
  return (unsigned)(size_t)p;
}

// ---------------------------------------------------------------------------
// Generic fused WMMA GEMM:  D = epilogue(A[M,K] @ Wt[N,K]^T)
//   Block tile 256(M) x 64(N); 8 waves, wave tile 32x64 = 2x4 WMMA tiles.
//   aMode 0: A is a plain f16 row-major matrix (ld = Kdim)
//   aMode 1: A rows are edge features gathered on the fly:
//            row r = (batch, point i, neighbor k);  col c < C -> h_i[c]
//            C <= c < 2C -> h_j[c-C] - h_i[c-C];  c >= 2C -> 0
//   epiMode 0: y = relu((acc + bias - mean)*gamma*rsqrt(var+eps) + beta) -> f16
//   epiMode 1: y = acc + bias -> f32           (shortcut projection)
//   epiMode 2: y = sq[m] + sq[n] - 2*acc -> f32 (pairwise sq distances, NT store)
// Batched via blockIdx.z with element strides (used for the per-batch Gram).
// ---------------------------------------------------------------------------
__global__ __launch_bounds__(256) void wmma_gemm_kernel(
    const _Float16* __restrict__ A, long long strideA,
    const _Float16* __restrict__ Wt, long long strideB,
    float* __restrict__ out32, _Float16* __restrict__ out16, long long strideOut,
    int M, int Kdim, int Nout,
    int aMode, const _Float16* __restrict__ gH, const int* __restrict__ gIdx,
    int gC, int gBatchBase,
    int epiMode,
    const float* __restrict__ pGamma, const float* __restrict__ pBeta,
    const float* __restrict__ pMean,  const float* __restrict__ pVar,
    const float* __restrict__ pBias,
    const float* __restrict__ pSq, int sqStride)
{
  __shared__ _Float16 sA[256 * LDS_STRIDE];
  __shared__ _Float16 sB[64 * LDS_STRIDE];

  const int z = blockIdx.z;
  const _Float16* Ab = A  + (long long)z * strideA;
  const _Float16* Wb = Wt + (long long)z * strideB;

  const int tid  = threadIdx.x;
  const int lane = tid & 31;
  const int wave = tid >> 5;        // 8 waves along M
  const int bm   = blockIdx.x * 256;
  const int bn   = blockIdx.y * 64;

  const v8f vzero = {0.f,0.f,0.f,0.f,0.f,0.f,0.f,0.f};
  v8f acc[2][4];
  #pragma unroll
  for (int mt = 0; mt < 2; ++mt)
    #pragma unroll
    for (int nt = 0; nt < 4; ++nt) acc[mt][nt] = vzero;

  const int br  = tid >> 2;          // B stage: 64 rows x 4 chunks
  const int bc0 = (tid & 3) * 8;
  const bool vecOK = ((Kdim & 7) == 0);

  for (int kk = 0; kk < Kdim; kk += 32) {
    // ---- stage A tile [256 x 32] into LDS (one row per thread) ----
    {
      const int gm = bm + tid;
      _Float16* dst = &sA[tid * LDS_STRIDE];
      if (aMode == 0) {
        const _Float16* src = Ab + (long long)gm * Kdim + kk;
        const bool rowOK = (bm < M);                 // uniform (M % 256 == 0)
        if (rowOK && vecOK && (kk + 32) <= Kdim) {   // uniform branch: async DMA
          #pragma unroll
          for (int q = 0; q < 4; ++q)
            async_copy_b128(lds_offset_of(dst + q * 8), src + q * 8);
          if (kk + 32 < Kdim) __builtin_prefetch(src + 32, 0, 0);
        } else {
          for (int c = 0; c < 32; ++c) {
            const int col = kk + c;
            dst[c] = (gm < M && col < Kdim) ? src[c] : (_Float16)0.f;
          }
        }
      } else {
        // fused edge gather: r = ((b*256 + i)*16 + k)
        const int k = gm & 15;
        const int i = (gm >> 4) & 255;
        const int b = gBatchBase + (gm >> 12);
        const int pI = b * 256 + i;
        const int j  = gIdx[(pI << 4) + k];
        const int pJ = b * 256 + j;
        const _Float16* hi = gH + (long long)pI * gC;
        const _Float16* hj = gH + (long long)pJ * gC;
        if ((gC & 15) == 0) {
          // uniform fast path (gC = 64/128): each 16-half chunk lies wholly
          // in the h_i half or wholly in the (h_j - h_i) half -> b128 loads
          // + packed f16 subtract.  (kk + 32 <= 2*gC always in this mode.)
          #pragma unroll
          for (int q = 0; q < 2; ++q) {
            const int col0 = kk + q * 16;            // uniform
            _Float16* d16 = dst + q * 16;
            if (col0 < gC) {
              *(v8h*)(d16)     = *(const v8h*)(hi + col0);
              *(v8h*)(d16 + 8) = *(const v8h*)(hi + col0 + 8);
            } else {
              const int cc = col0 - gC;
              const v8h j0 = *(const v8h*)(hj + cc);
              const v8h j1 = *(const v8h*)(hj + cc + 8);
              const v8h i0 = *(const v8h*)(hi + cc);
              const v8h i1 = *(const v8h*)(hi + cc + 8);
              *(v8h*)(d16)     = j0 - i0;
              *(v8h*)(d16 + 8) = j1 - i1;
            }
          }
        } else {
          // conv1 (gC = 3, Kdim = 6): tiny, scalar with zero pad
          for (int c = 0; c < 32; ++c) {
            const int col = kk + c;
            _Float16 v = (_Float16)0.f;
            if (col < gC)          v = hi[col];
            else if (col < 2 * gC) v = (_Float16)(hj[col - gC] - hi[col - gC]);
            dst[c] = v;
          }
        }
      }
    }
    // ---- stage B tile [64 x 32] (Wt is [Nout][Kdim], row n = channel) ----
    {
      const int gn = bn + br;
      const _Float16* srow = Wb + (long long)gn * Kdim;
      _Float16* dst = &sB[br * LDS_STRIDE + bc0];
      if (vecOK && (kk + bc0 + 8) <= Kdim) {         // uniform branch: async DMA
        async_copy_b128(lds_offset_of(dst), srow + kk + bc0);
      } else {
        #pragma unroll
        for (int c = 0; c < 8; ++c) {
          const int col = kk + bc0 + c;
          dst[c] = (col < Kdim) ? srow[col] : (_Float16)0.f;
        }
      }
    }
    wait_async_zero();        // drain this wave's ASYNCcnt before the barrier
    __syncthreads();

    // ---- build fragments per ISA 16-bit A/B lane layouts ----
    {
      const int lr = lane & 15;
      const int hs = lane >> 4;   // 0: K 0..7/16..23 (A and B); 1: other half
      v16h af[2], bf[4];
      #pragma unroll
      for (int mt = 0; mt < 2; ++mt) {
        const _Float16* p = &sA[(wave*32 + mt*16 + lr) * LDS_STRIDE + hs*8];
        const v8h lo = *(const v8h*)(p);
        const v8h hi = *(const v8h*)(p + 16);
        v16h a;
        #pragma unroll
        for (int t = 0; t < 8; ++t) { a[t] = lo[t]; a[t + 8] = hi[t]; }
        af[mt] = a;
      }
      #pragma unroll
      for (int nt = 0; nt < 4; ++nt) {
        const _Float16* p = &sB[(nt*16 + lr) * LDS_STRIDE + hs*8];
        const v8h lo = *(const v8h*)(p);
        const v8h hi = *(const v8h*)(p + 16);
        v16h b;
        #pragma unroll
        for (int t = 0; t < 8; ++t) { b[t] = lo[t]; b[t + 8] = hi[t]; }
        bf[nt] = b;
      }
      #pragma unroll
      for (int mt = 0; mt < 2; ++mt)
        #pragma unroll
        for (int nt = 0; nt < 4; ++nt)
          acc[mt][nt] = __builtin_amdgcn_wmma_f32_16x16x32_f16(
              false, af[mt], false, bf[nt], (short)0, acc[mt][nt], false, false);
    }
    __syncthreads();
  }

  // ---- epilogue: lane holds column n = lane&15, rows mBase..mBase+7 ----
  const int lr = lane & 15;
  const int hs = lane >> 4;
  #pragma unroll
  for (int mt = 0; mt < 2; ++mt) {
    #pragma unroll
    for (int nt = 0; nt < 4; ++nt) {
      const int n     = bn + nt*16 + lr;
      const int mBase = bm + wave*32 + mt*16 + hs*8;
      const v8f a = acc[mt][nt];
      if (epiMode == 0) {
        const float s  = pGamma[n] * rsqrtf(pVar[n] + BN_EPS_F);
        const float sh = (pBias[n] - pMean[n]) * s + pBeta[n];
        _Float16* o = out16 + (long long)z * strideOut;
        #pragma unroll
        for (int r = 0; r < 8; ++r) {
          const int m = mBase + r;
          if (m < M) o[(long long)m * Nout + n] = (_Float16)fmaxf(a[r]*s + sh, 0.f);
        }
      } else if (epiMode == 1) {
        const float bz = pBias[n];
        float* o = out32 + (long long)z * strideOut;
        #pragma unroll
        for (int r = 0; r < 8; ++r) {
          const int m = mBase + r;
          if (m < M) o[(long long)m * Nout + n] = a[r] + bz;
        }
      } else {
        const float* sqb = pSq + (long long)z * sqStride;
        const float sn = sqb[n];
        float* o = out32 + (long long)z * strideOut;
        #pragma unroll
        for (int r = 0; r < 8; ++r) {
          const int m = mBase + r;
          if (m < M)
            __builtin_nontemporal_store(sqb[m] + sn - 2.f * a[r],
                                        &o[(long long)m * Nout + n]);
        }
      }
    }
  }
}

// ---------------------------------------------------------------------------
// Helper kernels
// ---------------------------------------------------------------------------
__global__ void input_bn_kernel(const float* __restrict__ x,
                                const float* __restrict__ gamma,
                                const float* __restrict__ beta,
                                const float* __restrict__ mean,
                                const float* __restrict__ var,
                                float* __restrict__ H32, _Float16* __restrict__ H16) {
  const int p = blockIdx.x * 256 + threadIdx.x;
  if (p >= 32768) return;
  #pragma unroll
  for (int c = 0; c < 3; ++c) {
    const float v = (x[p*3 + c] - mean[c]) * (gamma[c] * rsqrtf(var[c] + BN_EPS_F)) + beta[c];
    H32[p*3 + c] = v;
    H16[p*3 + c] = (_Float16)v;
  }
}

__global__ void sq_kernel(const float* __restrict__ H, float* __restrict__ SQ,
                          int NP, int C) {
  const int p = blockIdx.x * 256 + threadIdx.x;
  if (p >= NP) return;
  const float* h = H + (long long)p * C;
  float s = 0.f;
  for (int c = 0; c < C; ++c) s += h[c] * h[c];
  SQ[p] = s;
}

__global__ void wt_convert_kernel(const float* __restrict__ W, _Float16* __restrict__ Wt,
                                  int Kdim, int Nout) {
  const int t = blockIdx.x * 256 + threadIdx.x;
  if (t >= Kdim * Nout) return;
  const int k = t / Nout, n = t % Nout;
  Wt[(long long)n * Kdim + k] = (_Float16)W[t];   // transpose to [Nout][Kdim]
}

// top-16 smallest per row of a [NP, 256] distance matrix (self included,
// matching the reference; neighbor order is irrelevant under max-pool)
__global__ void knn_kernel(const float* __restrict__ D, int* __restrict__ idx, int NP) {
  const int p = blockIdx.x * 256 + threadIdx.x;
  if (p >= NP) return;
  const float* d = D + (long long)p * 256;
  float best[16]; int bi[16];
  #pragma unroll
  for (int r = 0; r < 16; ++r) { best[r] = 3.4e38f; bi[r] = 0; }
  for (int j = 0; j < 256; ++j) {
    const float v = __builtin_nontemporal_load(&d[j]);
    float wv = best[0]; int wp = 0;
    #pragma unroll
    for (int r = 1; r < 16; ++r) { if (best[r] > wv) { wv = best[r]; wp = r; } }
    if (v < wv) {
      #pragma unroll
      for (int r = 0; r < 16; ++r) if (r == wp) { best[r] = v; bi[r] = j; }
    }
  }
  int* o = idx + p * 16;
  #pragma unroll
  for (int r = 0; r < 16; ++r) o[r] = bi[r];
}

// h_next[p, c] = max_k Y[(p_local*16 + k), c] + SC[p, c]
__global__ void max_shortcut_kernel(const _Float16* __restrict__ Y,
                                    const float* __restrict__ SC,
                                    float* __restrict__ H32, _Float16* __restrict__ H16,
                                    int ptBase, int nPts, int C) {
  const int t = blockIdx.x * 256 + threadIdx.x;
  if (t >= nPts * C) return;
  const int c  = t % C;
  const int pl = t / C;
  const int pg = ptBase + pl;
  float m = -3.4e38f;
  #pragma unroll
  for (int k = 0; k < 16; ++k)
    m = fmaxf(m, (float)Y[(long long)(pl * 16 + k) * C + c]);
  const float v = m + SC[(long long)pg * C + c];
  H32[(long long)pg * C + c] = v;
  H16[(long long)pg * C + c] = (_Float16)v;
}

__global__ void pool_kernel(const float* __restrict__ H, float* __restrict__ P) {
  const int b = blockIdx.x, c = threadIdx.x;
  const float* h = H + ((long long)b * 256) * 256 + c;
  float s = 0.f;
  for (int i = 0; i < 256; ++i) s += h[(long long)i * 256];
  P[b * 256 + c] = s * (1.f / 256.f);
}

__global__ void cls1_kernel(const float* __restrict__ P, const float* __restrict__ W,
                            const float* __restrict__ bia, float* __restrict__ Z) {
  const int t = blockIdx.x * 256 + threadIdx.x;   // 128*256
  const int b = t >> 8, o = t & 255;
  const float* p = P + b * 256;
  float s = bia[o];
  for (int k = 0; k < 256; ++k) s += p[k] * W[k * 256 + o];
  Z[t] = fmaxf(s, 0.f);
}

__global__ void cls2_kernel(const float* __restrict__ Z, const float* __restrict__ W,
                            const float* __restrict__ bia, float* __restrict__ out) {
  const int t = threadIdx.x;                       // 128*2 = 256
  const int b = t >> 1, o = t & 1;
  const float* z = Z + b * 256;
  float s = bia[o];
  for (int k = 0; k < 256; ++k) s += z[k] * W[k * 2 + o];
  out[t] = s;
}

// ---------------------------------------------------------------------------
// Orchestration. d_in leaf order = jax tree flatten (dict keys sorted):
//  0 cls1.W  1 cls1.b  2 cls2.W  3 cls2.b
//  4.. conv1 layers (W,b,bn.beta,bn.gamma,bn.mean,bn.var) x3  -> base 4
//  22.. conv2 x3 -> base 22 ; 40.. conv3 x3 -> base 40
//  58 input_bn.beta 59 .gamma 60 .mean 61 .var
//  62/63 sc1.W/b  64/65 sc2.W/b  66/67 sc3.W/b   68 x
// ---------------------------------------------------------------------------
extern "C" void kernel_launch(void* const* d_in, const int* in_sizes, int n_in,
                              void* d_out, int out_size, void* d_ws, size_t ws_size,
                              hipStream_t stream) {
  (void)in_sizes; (void)n_in; (void)out_size; (void)ws_size;
  const int B = 128, N = 256, NP = B * N;          // 32768 points
  auto F = [&](int i) { return (const float*)d_in[i]; };

  // workspace carve-up (deterministic every call)
  char* w = (char*)d_ws;
  size_t off = 0;
  auto alloc = [&](size_t bytes) { size_t r = off; off += (bytes + 255) & ~(size_t)255; return r; };
  _Float16* H16a = (_Float16*)(w + alloc((size_t)NP * 256 * 2));
  _Float16* H16b = (_Float16*)(w + alloc((size_t)NP * 256 * 2));
  float*    H32a = (float*)   (w + alloc((size_t)NP * 256 * 4));
  float*    H32b = (float*)   (w + alloc((size_t)NP * 256 * 4));
  float*    Dbuf = (float*)   (w + alloc((size_t)B * N * N * 4));
  int*      IDX  = (int*)     (w + alloc((size_t)NP * 16 * 4));
  float*    SQ   = (float*)   (w + alloc((size_t)NP * 4));
  _Float16* Y1   = (_Float16*)(w + alloc((size_t)65536 * 256 * 2));
  _Float16* Y2   = (_Float16*)(w + alloc((size_t)65536 * 256 * 2));
  float*    SC   = (float*)   (w + alloc((size_t)NP * 256 * 4));
  _Float16* Wt0  = (_Float16*)(w + alloc((size_t)256 * 256 * 2));
  _Float16* Wt1  = (_Float16*)(w + alloc((size_t)256 * 256 * 2));
  _Float16* Wt2  = (_Float16*)(w + alloc((size_t)256 * 256 * 2));
  _Float16* WtS  = (_Float16*)(w + alloc((size_t)256 * 256 * 2));
  float*    POOL = (float*)   (w + alloc((size_t)B * 256 * 4));
  float*    Z1   = (float*)   (w + alloc((size_t)B * 256 * 4));

  // input batch-norm -> h0 (f32 + f16)
  input_bn_kernel<<<NP / 256, 256, 0, stream>>>(F(68), F(59), F(58), F(60), F(61),
                                                H32a, H16a);

  const int convBase[3] = {4, 22, 40};
  const int scW[3] = {62, 64, 66};
  const int scB[3] = {63, 65, 67};
  const int Cin[3]  = {3, 64, 128};
  const int CoutA[3] = {64, 128, 256};

  _Float16* Hc16 = H16a; float* Hc32 = H32a;
  _Float16* Hn16 = H16b; float* Hn32 = H32b;

  for (int blk = 0; blk < 3; ++blk) {
    const int C = Cin[blk], Co = CoutA[blk], wb = convBase[blk];

    // per-point squared norms for the distance epilogue
    sq_kernel<<<NP / 256, 256, 0, stream>>>(Hc32, SQ, NP, C);

    // f16 transposed weights for the 3 MLP layers + shortcut
    const int kd0 = 2 * C;
    wt_convert_kernel<<<(kd0 * Co + 255) / 256, 256, 0, stream>>>(F(wb + 0),  Wt0, kd0, Co);
    wt_convert_kernel<<<(Co  * Co + 255) / 256, 256, 0, stream>>>(F(wb + 6),  Wt1, Co,  Co);
    wt_convert_kernel<<<(Co  * Co + 255) / 256, 256, 0, stream>>>(F(wb + 12), Wt2, Co,  Co);
    wt_convert_kernel<<<(C   * Co + 255) / 256, 256, 0, stream>>>(F(scW[blk]), WtS, C,  Co);

    // pairwise distances per batch:  D = sq_i + sq_j - 2 * H H^T   (WMMA)
    wmma_gemm_kernel<<<dim3(1, 4, B), 256, 0, stream>>>(
        Hc16, (long long)N * C, Hc16, (long long)N * C,
        Dbuf, nullptr, (long long)N * N,
        N, C, N,
        0, nullptr, nullptr, 0, 0,
        2, nullptr, nullptr, nullptr, nullptr, nullptr,
        SQ, N);

    // top-16 neighbors
    knn_kernel<<<NP / 256, 256, 0, stream>>>(Dbuf, IDX, NP);

    // shortcut projection: SC = H @ Wsc + bsc   (f32, whole point cloud)
    wmma_gemm_kernel<<<dim3(NP / 256, Co / 64, 1), 256, 0, stream>>>(
        Hc16, 0, WtS, 0,
        SC, nullptr, 0,
        NP, C, Co,
        0, nullptr, nullptr, 0, 0,
        1, nullptr, nullptr, nullptr, nullptr, F(scB[blk]),
        nullptr, 0);

    // edge MLP over 16-batch chunks (Mc = 16*256*16 = 65536 edge rows)
    const int Mc = 65536;
    for (int ch = 0; ch < 8; ++ch) {
      // layer 0: fused gather [h_i, h_j - h_i] -> BN+ReLU -> Y1
      wmma_gemm_kernel<<<dim3(Mc / 256, Co / 64, 1), 256, 0, stream>>>(
          nullptr, 0, Wt0, 0,
          nullptr, Y1, 0,
          Mc, 2 * C, Co,
          1, Hc16, IDX, C, ch * 16,
          0, F(wb + 3), F(wb + 2), F(wb + 4), F(wb + 5), F(wb + 1),
          nullptr, 0);
      // layer 1
      wmma_gemm_kernel<<<dim3(Mc / 256, Co / 64, 1), 256, 0, stream>>>(
          Y1, 0, Wt1, 0,
          nullptr, Y2, 0,
          Mc, Co, Co,
          0, nullptr, nullptr, 0, 0,
          0, F(wb + 9), F(wb + 8), F(wb + 10), F(wb + 11), F(wb + 7),
          nullptr, 0);
      // layer 2
      wmma_gemm_kernel<<<dim3(Mc / 256, Co / 64, 1), 256, 0, stream>>>(
          Y2, 0, Wt2, 0,
          nullptr, Y1, 0,
          Mc, Co, Co,
          0, nullptr, nullptr, 0, 0,
          0, F(wb + 15), F(wb + 14), F(wb + 16), F(wb + 17), F(wb + 13),
          nullptr, 0);
      // neighbor max-pool + shortcut add -> next block activations
      const int pts = 16 * N;  // 4096 points in this chunk
      max_shortcut_kernel<<<(pts * Co + 255) / 256, 256, 0, stream>>>(
          Y1, SC, Hn32, Hn16, ch * pts, pts, Co);
    }

    { _Float16* t16 = Hc16; Hc16 = Hn16; Hn16 = t16;
      float*    t32 = Hc32; Hc32 = Hn32; Hn32 = t32; }
  }

  // classifier head (tiny)
  pool_kernel<<<B, 256, 0, stream>>>(Hc32, POOL);
  cls1_kernel<<<(B * 256) / 256, 256, 0, stream>>>(POOL, F(0), F(1), Z1);
  cls2_kernel<<<1, 256, 0, stream>>>(Z1, F(2), F(3), (float*)d_out);
}
